// BahdanauAttention_15118284882237
// MI455X (gfx1250) — matrix operand order repaired and checked by reading, not compiled, not run on hardware
//
#include <hip/hip_runtime.h>

#define BB 8
#define LQ 512
#define LKC 512
#define DD 512
#define HH 128

typedef __attribute__((ext_vector_type(16))) _Float16 v16h;
typedef __attribute__((ext_vector_type(4)))  _Float16 v4h;
typedef __attribute__((ext_vector_type(8)))  float    v8f;

// gfx1250 hardware transcendental tanh (TRANS32 op)
__device__ __forceinline__ float fast_tanh(float x) {
    float r;
    asm volatile("v_tanh_f32 %0, %1" : "=v"(r) : "v"(x));
    return r;
}

// ---------------------------------------------------------------------------
// Kernel 1: Y(M,N) = X(M,K) * W(N,K)^T  (torch Linear layout), f16 WMMA.
// One wave per 16x64 output strip: A fragment loaded once per K-step,
// reused across 4 v_wmma issues (4 N-tiles).
// ---------------------------------------------------------------------------
__global__ void wmma_proj_kernel(const float* __restrict__ X,
                                 const float* __restrict__ W,
                                 float* __restrict__ Y,
                                 int M, int K, int N) {
    const int lane  = threadIdx.x & 31;
    const int lr    = lane & 15;
    const int hi    = lane >> 4;            // 0 = lanes 0-15, 1 = lanes 16-31
    const int mBase = blockIdx.x * 16;
    const int nGrp  = blockIdx.y * 64;

    const float* Xrow = X + (size_t)(mBase + lr) * K;

    v8f acc[4] = {};
    for (int kb = 0; kb < K; kb += 32) {
        // A 16x32 f16: element i -> K = kb + hi*16 + i
        v16h a;
#pragma unroll
        for (int i = 0; i < 8; ++i) {
            a[i]     = (_Float16)Xrow[kb + hi * 16 + i];
            a[i + 8] = (_Float16)Xrow[kb + hi * 16 + 8 + i];
        }
#pragma unroll
        for (int t = 0; t < 4; ++t) {
            // B 32x16 f16: element i -> K = kb + hi*16 + i, col n = nGrp+t*16+lr
            const float* Wrow = W + (size_t)(nGrp + t * 16 + lr) * K + kb + hi * 16;
            v16h bf;
#pragma unroll
            for (int i = 0; i < 16; ++i) bf[i] = (_Float16)Wrow[i];
            acc[t] = __builtin_amdgcn_wmma_f32_16x16x32_f16(false, a, false, bf,
                                                            (short)0, acc[t], false, false);
        }
    }
#pragma unroll
    for (int t = 0; t < 4; ++t) {
        float* Yt = Y + (size_t)(mBase + hi * 8) * N + nGrp + t * 16 + lr;
#pragma unroll
        for (int r = 0; r < 8; ++r) Yt[(size_t)r * N] = acc[t][r];
    }
}

// ---------------------------------------------------------------------------
// Kernel 2: LDS-tiled transpose + f32->f16 convert of value:
//   VT[b][n][k] = (f16) V[b][k][n]   -> contiguous-K operand for context WMMA
// 32x32 tiles, 256 threads, both global sides coalesced.
// ---------------------------------------------------------------------------
__global__ void value_cvt_transpose_kernel(const float* __restrict__ V,
                                           _Float16* __restrict__ VT) {
    const int b  = blockIdx.z;
    const int n0 = blockIdx.x * 32;
    const int k0 = blockIdx.y * 32;
    __shared__ _Float16 tile[32][36];       // 72B row stride: 8B-aligned, conflict-padded

    const int t  = threadIdx.x;             // 0..255
    const int r  = t >> 3;                  // 0..31
    const int c4 = (t & 7) * 4;             // 0,4,...,28

    const float4 f = *reinterpret_cast<const float4*>(
        V + ((size_t)b * LKC + (k0 + r)) * DD + n0 + c4);
    v4h hv = {(_Float16)f.x, (_Float16)f.y, (_Float16)f.z, (_Float16)f.w};
    *reinterpret_cast<v4h*>(&tile[r][c4]) = hv;
    __syncthreads();

    v4h o = {tile[c4 + 0][r], tile[c4 + 1][r], tile[c4 + 2][r], tile[c4 + 3][r]};
    *reinterpret_cast<v4h*>(
        VT + ((size_t)b * DD + (n0 + r)) * LKC + k0 + c4) = o;
}

// ---------------------------------------------------------------------------
// Kernel 3: scores + softmax. One block = one (b,q) row.
// Dominant cost of the whole problem: 268M hardware v_tanh_f32 (TRANS-bound;
// kp is L2-resident: 256KB/batch vs 192MB L2). Writes f32 attn to d_out and
// an f16 copy for the WMMA context GEMM.
// ---------------------------------------------------------------------------
__global__ void score_softmax_kernel(const float* __restrict__ qp,    // (B*LQ, H)
                                     const float* __restrict__ kp,    // (B*LK, H)
                                     const float* __restrict__ vvec,  // (H)
                                     float* __restrict__ attn,        // (B, LQ, LK)
                                     _Float16* __restrict__ attn_h) { // (B, LQ, LK)
    const int q = blockIdx.x;
    const int b = blockIdx.y;
    const int tid = threadIdx.x;            // 256 threads = 8 waves

    __shared__ __align__(16) float s_qp[HH];
    __shared__ __align__(16) float s_v[HH];
    __shared__ float s_score[LKC];
    __shared__ float red[256];

    if (tid < HH) {
        s_qp[tid] = qp[(size_t)(b * LQ + q) * HH + tid];
        s_v[tid]  = vvec[tid];
    }
    __syncthreads();

    for (int k = tid; k < LKC; k += 256) {
        const float* kr = kp + (size_t)(b * LKC + k) * HH;
        __builtin_prefetch(kr + HH, 0, 1);  // global_prefetch_b8
        float s = 0.f;
#pragma unroll 4
        for (int h = 0; h < HH; h += 4) {
            float4 kv = *reinterpret_cast<const float4*>(kr + h);
            float4 qv = *reinterpret_cast<const float4*>(&s_qp[h]);
            float4 vv = *reinterpret_cast<const float4*>(&s_v[h]);
            s += vv.x * fast_tanh(qv.x + kv.x);
            s += vv.y * fast_tanh(qv.y + kv.y);
            s += vv.z * fast_tanh(qv.z + kv.z);
            s += vv.w * fast_tanh(qv.w + kv.w);
        }
        s_score[k] = s;
    }
    __syncthreads();

    float m = -3.402823466e+38f;
    for (int k = tid; k < LKC; k += 256) m = fmaxf(m, s_score[k]);
    red[tid] = m;
    __syncthreads();
    for (int s = 128; s > 0; s >>= 1) {
        if (tid < s) red[tid] = fmaxf(red[tid], red[tid + s]);
        __syncthreads();
    }
    m = red[0];
    __syncthreads();

    float sum = 0.f;
    for (int k = tid; k < LKC; k += 256) {
        float e = __expf(s_score[k] - m);
        s_score[k] = e;
        sum += e;
    }
    red[tid] = sum;
    __syncthreads();
    for (int s = 128; s > 0; s >>= 1) {
        if (tid < s) red[tid] += red[tid + s];
        __syncthreads();
    }
    const float inv = 1.0f / red[0];

    float*     out  = attn   + (size_t)(b * LQ + q) * LKC;
    _Float16*  outh = attn_h + (size_t)(b * LQ + q) * LKC;
    for (int k = tid; k < LKC; k += 256) {
        float w = s_score[k] * inv;
        out[k]  = w;
        outh[k] = (_Float16)w;
    }
}

// ---------------------------------------------------------------------------
// Kernel 4: context = attn(b) [LQ x LK] * value(b) [LK x DV], f16 WMMA.
// One wave per 16x64 strip; A from f16 attn copy (contiguous 16B chunks),
// B from transposed f16 value (one 32B contiguous vector per fragment).
// ---------------------------------------------------------------------------
__global__ void wmma_context_kernel(const _Float16* __restrict__ AttnH,
                                    const _Float16* __restrict__ VT,
                                    float* __restrict__ Ctx) {
    const int b     = blockIdx.z;
    const int lane  = threadIdx.x & 31;
    const int lr    = lane & 15;
    const int hi    = lane >> 4;
    const int mBase = blockIdx.x * 16;
    const int nGrp  = blockIdx.y * 64;

    const _Float16* Arow = AttnH + ((size_t)b * LQ + mBase + lr) * LKC;
    float*          Cb   = Ctx + (size_t)b * LQ * DD;

    v8f acc[4] = {};
    for (int kb = 0; kb < LKC; kb += 32) {
        v16h a;
#pragma unroll
        for (int i = 0; i < 8; ++i) {       // two contiguous 16B chunks
            a[i]     = Arow[kb + hi * 16 + i];
            a[i + 8] = Arow[kb + hi * 16 + 8 + i];
        }
#pragma unroll
        for (int t = 0; t < 4; ++t) {
            const _Float16* Bp = VT + ((size_t)b * DD + nGrp + t * 16 + lr) * LKC
                                    + kb + hi * 16;
            v16h bf = *reinterpret_cast<const v16h*>(Bp);   // 32B contiguous
            acc[t] = __builtin_amdgcn_wmma_f32_16x16x32_f16(false, a, false, bf,
                                                            (short)0, acc[t], false, false);
        }
    }
#pragma unroll
    for (int t = 0; t < 4; ++t) {
        float* Ct = Cb + (size_t)(mBase + hi * 8) * DD + nGrp + t * 16 + lr;
#pragma unroll
        for (int r = 0; r < 8; ++r) Ct[(size_t)r * DD] = acc[t][r];
    }
}

// ---------------------------------------------------------------------------
extern "C" void kernel_launch(void* const* d_in, const int* in_sizes, int n_in,
                              void* d_out, int out_size, void* d_ws, size_t ws_size,
                              hipStream_t stream) {
    const float* query = (const float*)d_in[0];  // (B, LQ, D)
    const float* key   = (const float*)d_in[1];  // (B, LK, D)
    const float* value = (const float*)d_in[2];  // (B, LK, D)
    const float* w1    = (const float*)d_in[3];  // (H, D)
    const float* w2    = (const float*)d_in[4];  // (H, D)
    const float* vvec  = (const float*)d_in[5];  // (1, H)

    float* attn = (float*)d_out;                          // (B, LQ, LK)
    float* ctx  = attn + (size_t)BB * LQ * LKC;           // (B, LQ, DV)

    // workspace layout (all 32B aligned): qp 2MB | kp 2MB | attn_h 4MB | valT 4MB
    char* ws = (char*)d_ws;
    float*    qp     = (float*)(ws);
    float*    kp     = (float*)(ws + (2u << 20));
    _Float16* attn_h = (_Float16*)(ws + (4u << 20));
    _Float16* valT   = (_Float16*)(ws + (8u << 20));

    // Projections: M = 4096, K = 512, N = 128 (N/64 = 2 strip columns)
    dim3 pgrid(BB * LQ / 16, HH / 64);
    wmma_proj_kernel<<<pgrid, 32, 0, stream>>>(query, w1, qp, BB * LQ, DD, HH);
    wmma_proj_kernel<<<pgrid, 32, 0, stream>>>(key,   w2, kp, BB * LKC, DD, HH);

    // Transpose+convert value for the context GEMM (overlaps nothing it needs)
    value_cvt_transpose_kernel<<<dim3(DD / 32, LKC / 32, BB), 256, 0, stream>>>(value, valT);

    // Scores + softmax (dominant cost: 268M v_tanh_f32)
    score_softmax_kernel<<<dim3(LQ, BB), 256, 0, stream>>>(qp, kp, vvec, attn, attn_h);

    // Context GEMM per batch: 512x512x512, f16 WMMA
    wmma_context_kernel<<<dim3(LQ / 16, DD / 64, BB), 32, 0, stream>>>(attn_h, valT, ctx);
}
